// SelfAttentionBlock_39565238731439
// MI455X (gfx1250) — matrix operand order repaired and checked by reading, not compiled, not run on hardware
//
#include <hip/hip_runtime.h>
#include <hip/hip_bf16.h>
#include <math.h>

// ---------------- types ----------------
typedef __bf16 bf16_t;
typedef __attribute__((ext_vector_type(16))) __bf16 v16bf;
typedef __attribute__((ext_vector_type(8)))  __bf16 v8bf;
typedef __attribute__((ext_vector_type(8)))  float   v8f;
typedef v8bf __attribute__((may_alias)) v8bf_a;
// Explicit global-address-space pointer: survives the opaque asm below with
// its address space intact, so Q loads stay global_load (not flat_load).
typedef const bf16_t __attribute__((address_space(1)))* gbf16_cp;
typedef const v8bf_a __attribute__((address_space(1)))* v8bf_gcp;

// ---------------- problem dims ----------------
constexpr int BB   = 4;
constexpr int CIN  = 512;
constexpr int NPIX = 64 * 64;   // 4096
constexpr int CT   = 256;
constexpr int COUT = 512;

// ---------------- WMMA helpers ----------------
__device__ __forceinline__ v8f wmma_bf16(v16bf a, v16bf b, v8f c) {
  // 8 args: (neg_a, A, neg_b, B, c_mod, C, reuse_a, reuse_b)
  return __builtin_amdgcn_wmma_f32_16x16x32_bf16(false, a, false, b, (short)0, c,
                                                 false, false);
}

__device__ __forceinline__ v16bf frag_cat(v8bf a, v8bf b) {
  v16bf r;
#pragma unroll
  for (int i = 0; i < 8; ++i) { r[i] = a[i]; r[i + 8] = b[i]; }
  return r;
}

// A-matrix 16x32 bf16 fragment from row-major [row, K] storage.
// Lane l holds row (l&15); K segments: [k0+(l>>4)*8, +8) and that +16.
__device__ __forceinline__ v16bf load_fragA(const bf16_t* base, int ld, int lane,
                                            int k0) {
  const bf16_t* p = base + (size_t)(lane & 15) * ld + k0 + ((lane >> 4) << 3);
  v8bf lo = *(const v8bf_a*)p;
  v8bf hi = *(const v8bf_a*)(p + 16);
  return frag_cat(lo, hi);
}

// Same A-fragment load from an explicit global-AS pointer.
__device__ __forceinline__ v16bf load_fragA_g(gbf16_cp base, int ld, int lane,
                                              int k0) {
  gbf16_cp p = base + (size_t)(lane & 15) * ld + k0 + ((lane >> 4) << 3);
  v8bf lo = *(v8bf_gcp)p;
  v8bf hi = *(v8bf_gcp)(p + 16);
  return frag_cat(lo, hi);
}

// B-matrix 32x16 bf16 fragment. Storage is "B-transposed" row-major:
// row n holds K contiguous. Lane l holds column (l&15);
// K segments: [k0+(l>>4)*8, +8) and that +16.
__device__ __forceinline__ v16bf load_fragB(const bf16_t* base, int ld, int lane,
                                            int k0) {
  const bf16_t* p = base + (size_t)(lane & 15) * ld + k0 + ((lane >> 4) << 3);
  v8bf lo = *(const v8bf_a*)p;
  v8bf hi = *(const v8bf_a*)(p + 16);
  return frag_cat(lo, hi);
}

// ---------------- BN fold: W'[o,i] = W*scale[o] (bf16), bias'[o] ----------------
__global__ void fold_weight_kernel(const float* __restrict__ W,
                                   const float* __restrict__ gamma,
                                   const float* __restrict__ beta,
                                   const float* __restrict__ mean,
                                   const float* __restrict__ var,
                                   bf16_t* __restrict__ Wout,
                                   float* __restrict__ bout, int M, int K) {
  int idx = blockIdx.x * blockDim.x + threadIdx.x;
  if (idx >= M * K) return;
  int o = idx / K;
  int i = idx - o * K;
  float s = gamma[o] * rsqrtf(var[o] + 1e-5f);
  Wout[idx] = (bf16_t)(W[idx] * s);
  if (i == 0) bout[o] = beta[o] - mean[o] * s;
}

// ---------------- transpose+convert: f32 [B,Cin,N] -> bf16 [B,N,Cin] ----------------
__global__ void transpose_convert_kernel(const float* __restrict__ x,
                                         bf16_t* __restrict__ xt) {
  __shared__ float tile[32][33];
  const int b = blockIdx.z;
  const int n0 = blockIdx.x * 32, c0 = blockIdx.y * 32;
  const float* xb = x + (size_t)b * CIN * NPIX;
  bf16_t* xtb = xt + (size_t)b * NPIX * CIN;
  const int tx = threadIdx.x, ty = threadIdx.y;  // 32 x 8
#pragma unroll
  for (int j = 0; j < 4; ++j)
    tile[ty + 8 * j][tx] = xb[(size_t)(c0 + ty + 8 * j) * NPIX + n0 + tx];
  __syncthreads();
#pragma unroll
  for (int j = 0; j < 4; ++j)
    xtb[(size_t)(n0 + ty + 8 * j) * CIN + c0 + tx] = (bf16_t)tile[tx][ty + 8 * j];
}

// ---------------- generic WMMA GEMM: out = act(A[M,K] x B^T[N,K] + bias) ---------
// MODE 0: store bf16 transposed [Nn, M]   (Q^T, K^T)
// MODE 1: store bf16 normal     [M, Nn]   (V)
// MODE 2: store f32  normal     [M, Nn]   (final output)
template <int MODE>
__global__ void gemm_bias_relu_kernel(const bf16_t* __restrict__ A,
                                      const bf16_t* __restrict__ Bx,
                                      const float* __restrict__ bias,
                                      void* __restrict__ out, int M, int Nn,
                                      int K, size_t bstride_B,
                                      size_t bstride_out) {
  const int lane = threadIdx.x & 31;
  const int wv = threadIdx.x >> 5;
  const int b = blockIdx.z;
  const int m_base = blockIdx.x * 64 + wv * 16;
  const int n_base = blockIdx.y * 64;
  const bf16_t* Bb = Bx + (size_t)b * bstride_B;
  const bf16_t* Arow = A + (size_t)m_base * K;

  v8f acc[4];
#pragma unroll
  for (int t = 0; t < 4; ++t) acc[t] = (v8f){0.f, 0.f, 0.f, 0.f, 0.f, 0.f, 0.f, 0.f};

  for (int kc = 0; kc < K; kc += 32) {
    v16bf a = load_fragA(Arow, K, lane, kc);
#pragma unroll
    for (int t = 0; t < 4; ++t) {
      v16bf bf = load_fragB(Bb + (size_t)(n_base + t * 16) * K, K, lane, kc);
      acc[t] = wmma_bf16(a, bf, acc[t]);
    }
  }

  const int g = lane >> 4, nl = lane & 15;
  float bv[8];
#pragma unroll
  for (int r = 0; r < 8; ++r) bv[r] = bias[m_base + 8 * g + r];

  if (MODE == 0) {
    bf16_t* op = (bf16_t*)out + (size_t)b * bstride_out;
#pragma unroll
    for (int t = 0; t < 4; ++t) {
      const int n = n_base + t * 16 + nl;
      v8bf vec;
#pragma unroll
      for (int r = 0; r < 8; ++r) vec[r] = (bf16_t)fmaxf(acc[t][r] + bv[r], 0.f);
      *(v8bf_a*)(op + (size_t)n * M + m_base + 8 * g) = vec;
    }
  } else if (MODE == 1) {
    bf16_t* op = (bf16_t*)out + (size_t)b * bstride_out;
#pragma unroll
    for (int t = 0; t < 4; ++t)
#pragma unroll
      for (int r = 0; r < 8; ++r)
        op[(size_t)(m_base + r + 8 * g) * Nn + n_base + t * 16 + nl] =
            (bf16_t)fmaxf(acc[t][r] + bv[r], 0.f);
  } else {
    float* op = (float*)out + (size_t)b * bstride_out;
#pragma unroll
    for (int t = 0; t < 4; ++t)
#pragma unroll
      for (int r = 0; r < 8; ++r)
        op[(size_t)(m_base + r + 8 * g) * Nn + n_base + t * 16 + nl] =
            fmaxf(acc[t][r] + bv[r], 0.f);
  }
}

// ---------------- flash attention ----------------
// Qt, Kt: bf16 [B, N, Ct] ; Vm: bf16 [B, Ct, N] ; Ctxt out: bf16 [B, N, Ct]
// 4 waves / WG (64 query rows); all waves share K/V tiles staged in LDS.
__global__ void __launch_bounds__(128)
flash_attn_kernel(const bf16_t* __restrict__ Qt, const bf16_t* __restrict__ Kt,
                  const bf16_t* __restrict__ Vm, bf16_t* __restrict__ Ctxt) {
  const int tid = threadIdx.x;
  const int lane = tid & 31;
  const int wv = tid >> 5;
  const int b = blockIdx.y;
  const int q0 = blockIdx.x * 64 + wv * 16;
  const bf16_t* Qb = Qt + (size_t)b * NPIX * CT + (size_t)q0 * CT;
  const bf16_t* Kb = Kt + (size_t)b * NPIX * CT;
  const bf16_t* Vb = Vm + (size_t)b * CT * NPIX;
  const int g = lane >> 4, nl = lane & 15;

  __shared__ bf16_t Klds[32][CT];    // 16 KB: K rows kb..kb+31
  __shared__ bf16_t Vlds[CT][32];    // 16 KB: V cols c=0..255, m rel 0..31
  __shared__ bf16_t Pl[4][16][32];   //  4 KB: per-wave P bounce

  v8f O[16];
#pragma unroll
  for (int t = 0; t < 16; ++t) O[t] = (v8f){0.f, 0.f, 0.f, 0.f, 0.f, 0.f, 0.f, 0.f};
  float m_r[8], l_r[8];
#pragma unroll
  for (int r = 0; r < 8; ++r) { m_r[r] = -INFINITY; l_r[r] = 0.f; }

  // Carry the Q base address through an opaque asm each iteration so LICM
  // cannot hoist the (loop-invariant) Q fragment loads out of the key loop
  // and spill them to scratch. The pointer keeps its addrspace(1) type, so
  // the loads remain global_load_b128 (LOADcnt only, no flat/LDS-pipe use).
  gbf16_cp q_ptr = (gbf16_cp)Qb;

  for (int kb = 0; kb < NPIX; kb += 32) {
    asm volatile("" : "+v"(q_ptr));

    // ---- cooperative staging of this key block's K and V tiles ----
    __syncthreads();  // previous iteration's LDS reads complete
    {
      // K rows kb..kb+31: 32*256 bf16, contiguous in memory (16 KB)
      const bf16_t* Ksrc = Kb + (size_t)kb * CT;
#pragma unroll
      for (int j = 0; j < 8; ++j) {
        const int idx = (tid + 128 * j) * 8;
        *(v8bf_a*)(&Klds[0][0] + idx) = *(const v8bf_a*)(Ksrc + idx);
      }
      // V: 256 rows (c), 32 m-contiguous each
      const int tx = tid & 3, rr = tid >> 2;
#pragma unroll
      for (int j = 0; j < 8; ++j) {
        const int c = rr + 32 * j;
        *(v8bf_a*)(&Vlds[c][tx * 8]) =
            *(const v8bf_a*)(Vb + (size_t)c * NPIX + kb + tx * 8);
      }
      // prefetch next key block into caches while we compute this one
      if (kb + 32 < NPIX) {
        __builtin_prefetch(Kb + (size_t)(kb + 32) * CT + tid * 64, 0, 0);
        __builtin_prefetch(Vb + (size_t)(tid * 2) * NPIX + kb + 32, 0, 0);
      }
    }
    __syncthreads();

    // ---- S = Q K^T over this 32-key block (two 16x16 tiles) ----
    v8f S0 = (v8f){0.f, 0.f, 0.f, 0.f, 0.f, 0.f, 0.f, 0.f};
    v8f S1 = (v8f){0.f, 0.f, 0.f, 0.f, 0.f, 0.f, 0.f, 0.f};
#pragma unroll
    for (int i = 0; i < 8; ++i) {
      v16bf qa = load_fragA_g(q_ptr, CT, lane, i * 32);  // stays in-loop (L0 hit)
      v16bf k0f = load_fragB(&Klds[0][0], CT, lane, i * 32);
      v16bf k1f = load_fragB(&Klds[16][0], CT, lane, i * 32);
      S0 = wmma_bf16(qa, k0f, S0);
      S1 = wmma_bf16(qa, k1f, S1);
    }

    // ---- online softmax; row stats align with D-layout (row = r + 8*(lane>>4)) ----
    float p0[8], p1[8], alpha[8];
#pragma unroll
    for (int r = 0; r < 8; ++r) {
      float s0 = S0[r] * 0.0625f;  // Ct^-0.5 = 1/16
      float s1 = S1[r] * 0.0625f;
      float mx = fmaxf(s0, s1);
      mx = fmaxf(mx, __shfl_xor(mx, 1, 32));
      mx = fmaxf(mx, __shfl_xor(mx, 2, 32));
      mx = fmaxf(mx, __shfl_xor(mx, 4, 32));
      mx = fmaxf(mx, __shfl_xor(mx, 8, 32));
      float mn = fmaxf(m_r[r], mx);
      alpha[r] = __expf(m_r[r] - mn);
      m_r[r] = mn;
      p0[r] = __expf(s0 - mn);
      p1[r] = __expf(s1 - mn);
      float sm = p0[r] + p1[r];
      sm += __shfl_xor(sm, 1, 32);
      sm += __shfl_xor(sm, 2, 32);
      sm += __shfl_xor(sm, 4, 32);
      sm += __shfl_xor(sm, 8, 32);
      l_r[r] = l_r[r] * alpha[r] + sm;
    }
#pragma unroll
    for (int t = 0; t < 16; ++t)
#pragma unroll
      for (int r = 0; r < 8; ++r) O[t][r] *= alpha[r];

    // ---- P: D-layout -> LDS -> A-layout (same-wave DS ops are in order) ----
#pragma unroll
    for (int r = 0; r < 8; ++r) {
      Pl[wv][r + 8 * g][nl] = (bf16_t)p0[r];
      Pl[wv][r + 8 * g][16 + nl] = (bf16_t)p1[r];
    }
    asm volatile("s_wait_dscnt 0x0" ::: "memory");
    v16bf pa = load_fragA(&Pl[wv][0][0], 32, lane, 0);

    // ---- O += P V (V fragments from LDS) ----
#pragma unroll
    for (int ct = 0; ct < 16; ++ct) {
      v16bf vbf = load_fragB(&Vlds[ct * 16][0], 32, lane, 0);
      O[ct] = wmma_bf16(pa, vbf, O[ct]);
    }
  }

  // ---- finalize: divide by row sums, store ctx^T [N, Ct] bf16 ----
  float inv[8];
#pragma unroll
  for (int r = 0; r < 8; ++r) inv[r] = 1.0f / l_r[r];
  bf16_t* Cb = Ctxt + (size_t)b * NPIX * CT + (size_t)q0 * CT;
#pragma unroll
  for (int ct = 0; ct < 16; ++ct)
#pragma unroll
    for (int r = 0; r < 8; ++r)
      Cb[(size_t)(r + 8 * g) * CT + ct * 16 + nl] = (bf16_t)(O[ct][r] * inv[r]);
}

// ---------------- host-side launcher ----------------
extern "C" void kernel_launch(void* const* d_in, const int* in_sizes, int n_in,
                              void* d_out, int out_size, void* d_ws,
                              size_t ws_size, hipStream_t stream) {
  const float* query_feats = (const float*)d_in[0];
  const float* key_feats = (const float*)d_in[1];
  const float* Wq = (const float*)d_in[2];
  const float* gq = (const float*)d_in[3];
  const float* bq = (const float*)d_in[4];
  const float* mq = (const float*)d_in[5];
  const float* vq = (const float*)d_in[6];
  const float* Wk = (const float*)d_in[7];
  const float* gk = (const float*)d_in[8];
  const float* bk = (const float*)d_in[9];
  const float* mk = (const float*)d_in[10];
  const float* vk = (const float*)d_in[11];
  const float* Wv = (const float*)d_in[12];
  const float* gv = (const float*)d_in[13];
  const float* bv = (const float*)d_in[14];
  const float* mv = (const float*)d_in[15];
  const float* vv = (const float*)d_in[16];
  const float* Wo = (const float*)d_in[17];
  const float* go = (const float*)d_in[18];
  const float* bo = (const float*)d_in[19];
  const float* mo = (const float*)d_in[20];
  const float* vo = (const float*)d_in[21];

  uint8_t* ws = (uint8_t*)d_ws;
  size_t off = 0;
  auto alloc = [&](size_t bytes) {
    size_t r = off;
    off = (off + bytes + 255) & ~(size_t)255;
    return r;
  };
  bf16_t* Xq_t = (bf16_t*)(ws + alloc((size_t)BB * NPIX * CIN * 2));
  bf16_t* Xk_t = (bf16_t*)(ws + alloc((size_t)BB * NPIX * CIN * 2));
  bf16_t* Wq_b = (bf16_t*)(ws + alloc((size_t)CT * CIN * 2));
  bf16_t* Wk_b = (bf16_t*)(ws + alloc((size_t)CT * CIN * 2));
  bf16_t* Wv_b = (bf16_t*)(ws + alloc((size_t)CT * CIN * 2));
  bf16_t* Wo_b = (bf16_t*)(ws + alloc((size_t)COUT * CT * 2));
  float* biasq = (float*)(ws + alloc(CT * 4));
  float* biask = (float*)(ws + alloc(CT * 4));
  float* biasv = (float*)(ws + alloc(CT * 4));
  float* biaso = (float*)(ws + alloc(COUT * 4));
  bf16_t* Qt = (bf16_t*)(ws + alloc((size_t)BB * NPIX * CT * 2));
  bf16_t* Kt = (bf16_t*)(ws + alloc((size_t)BB * NPIX * CT * 2));
  bf16_t* Vm = (bf16_t*)(ws + alloc((size_t)BB * CT * NPIX * 2));
  bf16_t* Ctxt = (bf16_t*)(ws + alloc((size_t)BB * NPIX * CT * 2));

  // 1) fold BN into weights
  {
    int n1 = CT * CIN, n2 = COUT * CT;
    fold_weight_kernel<<<(n1 + 255) / 256, 256, 0, stream>>>(Wq, gq, bq, mq, vq,
                                                             Wq_b, biasq, CT, CIN);
    fold_weight_kernel<<<(n1 + 255) / 256, 256, 0, stream>>>(Wk, gk, bk, mk, vk,
                                                             Wk_b, biask, CT, CIN);
    fold_weight_kernel<<<(n1 + 255) / 256, 256, 0, stream>>>(Wv, gv, bv, mv, vv,
                                                             Wv_b, biasv, CT, CIN);
    fold_weight_kernel<<<(n2 + 255) / 256, 256, 0, stream>>>(Wo, go, bo, mo, vo,
                                                             Wo_b, biaso, COUT, CT);
  }

  // 2) transpose + bf16 convert inputs
  {
    dim3 tb(32, 8);
    dim3 tg(NPIX / 32, CIN / 32, BB);
    transpose_convert_kernel<<<tg, tb, 0, stream>>>(query_feats, Xq_t);
    transpose_convert_kernel<<<tg, tb, 0, stream>>>(key_feats, Xk_t);
  }

  // 3) projections (bf16 WMMA GEMM + bias + ReLU)
  {
    dim3 gp(CT / 64, NPIX / 64, BB);
    gemm_bias_relu_kernel<0><<<gp, 128, 0, stream>>>(
        Wq_b, Xq_t, biasq, Qt, CT, NPIX, CIN, (size_t)NPIX * CIN,
        (size_t)NPIX * CT);
    gemm_bias_relu_kernel<0><<<gp, 128, 0, stream>>>(
        Wk_b, Xk_t, biask, Kt, CT, NPIX, CIN, (size_t)NPIX * CIN,
        (size_t)NPIX * CT);
    gemm_bias_relu_kernel<1><<<gp, 128, 0, stream>>>(
        Wv_b, Xk_t, biasv, Vm, CT, NPIX, CIN, (size_t)NPIX * CIN,
        (size_t)CT * NPIX);
  }

  // 4) flash attention (softmax(QK^T/16) V), never materializes N x N
  {
    dim3 fg(NPIX / 64, BB);
    flash_attn_kernel<<<fg, 128, 0, stream>>>(Qt, Kt, Vm, Ctxt);
  }

  // 5) output projection + BN + ReLU -> f32 d_out [B, Cout, N]
  {
    dim3 go_(COUT / 64, NPIX / 64, BB);
    gemm_bias_relu_kernel<2><<<go_, 128, 0, stream>>>(
        Wo_b, Ctxt, biaso, d_out, COUT, NPIX, CT, (size_t)NPIX * CT,
        (size_t)COUT * NPIX);
  }
}
